// GroundEncoder_73057393705695
// MI455X (gfx1250) — matrix operand order repaired and checked by reading, not compiled, not run on hardware
//
#include <hip/hip_runtime.h>
#include <hip/hip_bf16.h>
#include <math.h>

typedef __attribute__((ext_vector_type(16))) _Float16 v16h;
typedef __attribute__((ext_vector_type(4)))  _Float16 h4;
typedef __attribute__((ext_vector_type(8)))  float    v8f;

#define BM 128
#define BN 128
#define BK 32
#define LDA (BK + 8)
#define LDB (BK + 8)
#define MPAD 1664   // 13 * 128, padded row count for all activation buffers

// ---------------------------------------------------------------- utilities

__device__ __forceinline__ float gelu_exact(float x) {
    return 0.5f * x * (1.0f + erff(x * 0.70710678118654752f));
}

__device__ __forceinline__ void atomicMaxFloat(float* addr, float val) {
    int* ai = (int*)addr;
    int old = __float_as_int(*addr);
    while (__int_as_float(old) < val) {
        int assumed = old;
        old = atomicCAS(ai, assumed, __float_as_int(val));
        if (old == assumed) break;
    }
}

// ---------------------------------------------------------------- WMMA GEMM
// C[., N] = act(A[., K] @ B[K, N] + bias) (+ resid), fp32 in HBM, f16 WMMA
// with fp32 accumulate. All row dimensions are padded to MPAD so the fill and
// epilogue are completely branch-free; K % 32 == 0, N % 128 == 0 at every
// call site. Global->LDS fill is float4 (global_load_b128) with register
// double-buffering so loads overlap the 8 WMMAs per K-step.
__global__ __launch_bounds__(256) void gemm_wmma_kernel(
    const float* __restrict__ A, const float* __restrict__ Bw,
    const float* __restrict__ bias, const float* __restrict__ resid,
    float* __restrict__ C, int N, int K, int act)
{
    __shared__ _Float16 As[BM][LDA];   // [m][k]
    __shared__ _Float16 Bs[BN][LDB];   // [n][k]  (transposed on fill)

    const int tid  = threadIdx.x;
    const int wave = tid >> 5;
    const int lane = tid & 31;
    const int bm = blockIdx.y * BM;
    const int bn = blockIdx.x * BN;
    const int wm = (wave >> 1) * 32;   // 4 wave-rows of 32
    const int wn = (wave & 1) * 64;    // 2 wave-cols of 64
    const int mrow = lane & 15;
    const int hi   = lane >> 4;        // half-wave select

    v8f acc[2][4];
    v8f zero = {};
#pragma unroll
    for (int t = 0; t < 2; ++t)
#pragma unroll
        for (int u = 0; u < 4; ++u) acc[t][u] = zero;

    // per-thread fill coordinates (4 float4 chunks per tile per matrix)
    int aR[4], aC[4], bR[4], bC[4];
#pragma unroll
    for (int it = 0; it < 4; ++it) {
        int e = tid * 4 + it * 1024;
        aR[it] = e >> 5;  aC[it] = e & 31;    // A: 128 x 32
        bR[it] = e >> 7;  bC[it] = e & 127;   // B: 32 x 128
    }

    float4 aStage[4], bStage[4];

    // prologue: stage first tile
#pragma unroll
    for (int it = 0; it < 4; ++it) {
        aStage[it] = *reinterpret_cast<const float4*>(
            &A[(size_t)(bm + aR[it]) * K + aC[it]]);
        bStage[it] = *reinterpret_cast<const float4*>(
            &Bw[(size_t)bR[it] * N + bn + bC[it]]);
    }

    for (int k0 = 0; k0 < K; k0 += BK) {
        // commit staged tile to LDS (A as packed 4-half vector stores)
#pragma unroll
        for (int it = 0; it < 4; ++it) {
            h4 av = { (_Float16)aStage[it].x, (_Float16)aStage[it].y,
                      (_Float16)aStage[it].z, (_Float16)aStage[it].w };
            *reinterpret_cast<h4*>(&As[aR[it]][aC[it]]) = av;
            Bs[bC[it] + 0][bR[it]] = (_Float16)bStage[it].x;
            Bs[bC[it] + 1][bR[it]] = (_Float16)bStage[it].y;
            Bs[bC[it] + 2][bR[it]] = (_Float16)bStage[it].z;
            Bs[bC[it] + 3][bR[it]] = (_Float16)bStage[it].w;
        }
        __syncthreads();

        // stage next tile while WMMAs run
        int kn = k0 + BK;
        if (kn < K) {
#pragma unroll
            for (int it = 0; it < 4; ++it) {
                aStage[it] = *reinterpret_cast<const float4*>(
                    &A[(size_t)(bm + aR[it]) * K + kn + aC[it]]);
                bStage[it] = *reinterpret_cast<const float4*>(
                    &Bw[(size_t)(kn + bR[it]) * N + bn + bC[it]]);
            }
        }

        // fragments per ISA 7.12.2 layouts
        v16h afrag[2], bfrag[4];
#pragma unroll
        for (int t = 0; t < 2; ++t) {
            const _Float16* ap = &As[wm + t * 16 + mrow][0];
            int b0 = hi * 8;
#pragma unroll
            for (int i = 0; i < 8; ++i) afrag[t][i]     = ap[b0 + i];
#pragma unroll
            for (int i = 0; i < 8; ++i) afrag[t][8 + i] = ap[16 + b0 + i];
        }
#pragma unroll
        for (int u = 0; u < 4; ++u) {
            const _Float16* bp = &Bs[wn + u * 16 + mrow][0];
#pragma unroll
            for (int i = 0; i < 8; ++i) bfrag[u][i]     = bp[hi * 8 + i];
#pragma unroll
            for (int i = 0; i < 8; ++i) bfrag[u][8 + i] = bp[16 + hi * 8 + i];
        }
#pragma unroll
        for (int t = 0; t < 2; ++t)
#pragma unroll
            for (int u = 0; u < 4; ++u)
                acc[t][u] = __builtin_amdgcn_wmma_f32_16x16x32_f16(
                    false, afrag[t], false, bfrag[u],
                    (short)0, acc[t][u], false, false);
        __syncthreads();
    }

    // branch-free epilogue: bias / act / residual
#pragma unroll
    for (int t = 0; t < 2; ++t) {
#pragma unroll
        for (int u = 0; u < 4; ++u) {
            int col = bn + wn + u * 16 + mrow;
            float bv = bias ? bias[col] : 0.0f;
#pragma unroll
            for (int r = 0; r < 8; ++r) {
                int row = bm + wm + t * 16 + r + hi * 8;
                float v = acc[t][u][r] + bv;
                if (act == 1) v = gelu_exact(v);
                size_t o = (size_t)row * N + col;
                if (resid) v += resid[o];
                C[o] = v;
            }
        }
    }
}

// ---------------------------------------------------------------- LayerNorm
__global__ __launch_bounds__(256) void ln_kernel(
    const float* __restrict__ x, const float* __restrict__ g,
    const float* __restrict__ b, float* __restrict__ y)
{
    int row = blockIdx.x, t = threadIdx.x;
    const float* xr = x + (size_t)row * 768;
    __shared__ float red[256];
    float s = 0.0f;
    for (int i = t; i < 768; i += 256) s += xr[i];
    red[t] = s; __syncthreads();
    for (int k = 128; k > 0; k >>= 1) { if (t < k) red[t] += red[t + k]; __syncthreads(); }
    float mean = red[0] * (1.0f / 768.0f);
    __syncthreads();
    float v = 0.0f;
    for (int i = t; i < 768; i += 256) { float d = xr[i] - mean; v += d * d; }
    red[t] = v; __syncthreads();
    for (int k = 128; k > 0; k >>= 1) { if (t < k) red[t] += red[t + k]; __syncthreads(); }
    float inv = rsqrtf(red[0] * (1.0f / 768.0f) + 1e-6f);
    for (int i = t; i < 768; i += 256)
        y[(size_t)row * 768 + i] = (xr[i] - mean) * inv * g[i] + b[i];
}

// ---------------------------------------------------------------- attention
// one block (64 threads) per (img, head, q-row); 197 keys, DH=64
__global__ __launch_bounds__(64) void attn_kernel(
    const float* __restrict__ qkv, float* __restrict__ out)
{
    int q = blockIdx.x, h = blockIdx.y, img = blockIdx.z;
    int t = threadIdx.x;
    __shared__ float qv[64];
    __shared__ float sc[197];
    __shared__ float red[64];
    const float* base = qkv + (size_t)img * 197 * 2304;

    qv[t] = base[(size_t)q * 2304 + h * 64 + t];           // Q slab (which=0)
    __syncthreads();
    for (int k = t; k < 197; k += 64) {
        const float* kp = base + (size_t)k * 2304 + (12 + h) * 64;   // K slab
        float s = 0.0f;
#pragma unroll
        for (int d = 0; d < 64; ++d) s += qv[d] * kp[d];
        sc[k] = s * 0.125f;                                 // 1/sqrt(64)
    }
    __syncthreads();
    float m = -1e30f;
    for (int k = t; k < 197; k += 64) m = fmaxf(m, sc[k]);
    red[t] = m; __syncthreads();
    for (int s = 32; s > 0; s >>= 1) { if (t < s) red[t] = fmaxf(red[t], red[t + s]); __syncthreads(); }
    m = red[0]; __syncthreads();
    float sum = 0.0f;
    for (int k = t; k < 197; k += 64) { float e = __expf(sc[k] - m); sc[k] = e; sum += e; }
    red[t] = sum; __syncthreads();
    for (int s = 32; s > 0; s >>= 1) { if (t < s) red[t] += red[t + s]; __syncthreads(); }
    float inv = 1.0f / red[0];
    __syncthreads();
    float o = 0.0f;
    for (int k = 0; k < 197; ++k)
        o += sc[k] * base[(size_t)k * 2304 + (24 + h) * 64 + t];     // V slab
    out[((size_t)img * 197 + q) * 768 + h * 64 + t] = o * inv;
}

// ---------------------------------------------------------------- patch path
__global__ __launch_bounds__(256) void im2col_kernel(
    const float* __restrict__ imgs, float* __restrict__ Ap)
{
    int p = blockIdx.x;                 // 0..1567 = img*196 + patch
    int img = p / 196, pi = p % 196;
    int hy = pi / 14, hx = pi % 14;
    for (int k = threadIdx.x; k < 768; k += 256) {
        int c = k >> 8, py = (k >> 4) & 15, px = k & 15;
        Ap[(size_t)p * 768 + k] =
            imgs[(((size_t)img * 3 + c) * 224 + hy * 16 + py) * 224 + hx * 16 + px];
    }
}

__global__ __launch_bounds__(256) void assemble_tok_kernel(
    const float* __restrict__ Ptok, const float* __restrict__ cls,
    const float* __restrict__ pos, float* __restrict__ tok)
{
    int r = blockIdx.x;                 // 0..1575 = img*197 + s
    int img = r / 197, s = r % 197;
    for (int c = threadIdx.x; c < 768; c += 256) {
        float v = (s == 0) ? cls[c] : Ptok[((size_t)img * 196 + (s - 1)) * 768 + c];
        tok[(size_t)r * 768 + c] = v + pos[(size_t)s * 768 + c];
    }
}

// ---------------------------------------------------------------- geometry
__global__ void inv4_kernel(const float* __restrict__ poses, float* __restrict__ c2w) {
    int i = threadIdx.x;
    if (i >= 8) return;
    const int perm[4] = {1, 0, 2, 3};
    float a[4][8];
    for (int r = 0; r < 4; ++r) {
        for (int c = 0; c < 4; ++c) a[r][c] = poses[i * 16 + perm[r] * 4 + c];
        for (int c = 0; c < 4; ++c) a[r][4 + c] = (r == c) ? 1.0f : 0.0f;
    }
    for (int c = 0; c < 4; ++c) {
        int p = c;
        for (int r = c + 1; r < 4; ++r) if (fabsf(a[r][c]) > fabsf(a[p][c])) p = r;
        if (p != c) for (int k = 0; k < 8; ++k) { float tm = a[c][k]; a[c][k] = a[p][k]; a[p][k] = tm; }
        float inv = 1.0f / a[c][c];
        for (int k = 0; k < 8; ++k) a[c][k] *= inv;
        for (int r = 0; r < 4; ++r) if (r != c) {
            float f = a[r][c];
            for (int k = 0; k < 8; ++k) a[r][k] -= f * a[c][k];
        }
    }
    for (int r = 0; r < 4; ++r)
        for (int c = 0; c < 4; ++c) c2w[i * 16 + r * 4 + c] = a[r][4 + c];
}

__global__ __launch_bounds__(256) void depth_resize_kernel(
    const float* __restrict__ depths, float* __restrict__ d14)
{
    int bn = blockIdx.x, t = threadIdx.x;
    if (t >= 196) return;
    int oy = t / 14, ox = t % 14;
    float sy = (oy + 0.5f) * 16.0f - 0.5f;
    float sx = (ox + 0.5f) * 16.0f - 0.5f;
    int y0 = (int)floorf(sy); float fy = sy - (float)y0;
    int x0 = (int)floorf(sx); float fx = sx - (float)x0;
    int y1 = min(y0 + 1, 223); y0 = max(y0, 0);
    int x1 = min(x0 + 1, 223); x0 = max(x0, 0);
    const float* dp = depths + (size_t)bn * 224 * 224;
    float v = (1.0f - fy) * ((1.0f - fx) * dp[y0 * 224 + x0] + fx * dp[y0 * 224 + x1])
            +         fy  * ((1.0f - fx) * dp[y1 * 224 + x0] + fx * dp[y1 * 224 + x1]);
    d14[bn * 196 + t] = v * 65.535f;   // * 65535/1000
}

// ---------------------------------------------------------------- BEV scatter
__global__ __launch_bounds__(256) void zero_fill_kernel(float* p, int n) {
    int i = blockIdx.x * 256 + threadIdx.x;
    if (i < n) p[i] = 0.0f;
}

__global__ __launch_bounds__(256) void scatter_kernel(
    const float* __restrict__ tok, const float* __restrict__ d14,
    const float* __restrict__ intr, const float* __restrict__ c2w,
    float* __restrict__ ssum, float* __restrict__ smax, float* __restrict__ cnt)
{
    int p = blockIdx.x;                 // 0..1567 = (b*4+n)*196 + pix
    int bn = p / 196, pix = p % 196;
    int b = bn / 4;
    int h = pix / 14, w = pix % 14;
    __shared__ int sidx, svalid;
    if (threadIdx.x == 0) {
        float d = d14[bn * 196 + pix];
        bool mask = (d > 0.0f) && (d < 5.0f);
        const float* Km = intr + bn * 9;
        float fx = Km[0], fy = Km[4], cx = Km[2], cy = Km[5];
        float Xc = -(((float)w - cx) * d / (fx + 1e-8f));
        float Yc =  (((float)h - cy) * d / (fy + 1e-8f));
        const float* M4 = c2w + bn * 16;
        float Xw = M4[0] * Xc + M4[1] * Yc + M4[2] * d + M4[3];
        float Yw = M4[4] * Xc + M4[5] * Yc + M4[6] * d + M4[7];
        float px = Xw * 1.4f + 7.0f;            // HF/tile = 1.4, HF/2 = 7
        float py = -(Yw * 1.4f) + 7.0f;
        int pxi = (int)rintf(px);               // round-half-even like jnp.round
        int pyi = (int)rintf(py);
        bool inb = pxi >= 0 && pxi < 14 && pyi >= 0 && pyi < 14;
        svalid = (mask && inb) ? 1 : 0;
        sidx = b * 196 + pyi * 14 + pxi;
    }
    __syncthreads();
    if (!svalid) return;
    int idx = sidx;
    const float* src = tok + ((size_t)bn * 197 + 1 + pix) * 768;  // skip CLS
    for (int c = threadIdx.x; c < 768; c += 256) {
        float v = src[c];
        atomicAdd(&ssum[(size_t)idx * 768 + c], v);
        atomicMaxFloat(&smax[(size_t)idx * 768 + c], v);   // zero-initialized -> max(0, .)
    }
    if (threadIdx.x == 0) atomicAdd(&cnt[idx], 1.0f);
}

// merge avg/max + 1x1 conv (768 -> 256)
__global__ __launch_bounds__(256) void bev_conv_kernel(
    const float* __restrict__ ssum, const float* __restrict__ smax,
    const float* __restrict__ cnt, const float* __restrict__ convw,
    const float* __restrict__ convb, float* __restrict__ out)
{
    int p = blockIdx.x;                 // 0..391 = b*196 + pix
    int b = p / 196, pix = p % 196;
    __shared__ float bev[768];
    float c_ = cnt[p];
    for (int c = threadIdx.x; c < 768; c += 256) {
        float a = ssum[(size_t)p * 768 + c] / (c_ + 1e-8f);
        bev[c] = 0.5f * (a + smax[(size_t)p * 768 + c]);
    }
    __syncthreads();
    int o = threadIdx.x;                // 256 output channels
    float acc = convb[o];
    for (int c = 0; c < 768; ++c) acc += bev[c] * convw[o * 768 + c];
    out[((size_t)b * 256 + o) * 196 + pix] = acc;
}

// ---------------------------------------------------------------- launcher
extern "C" void kernel_launch(void* const* d_in, const int* in_sizes, int n_in,
                              void* d_out, int out_size, void* d_ws, size_t ws_size,
                              hipStream_t stream) {
    const float* imgs    = (const float*)d_in[0];
    const float* depths  = (const float*)d_in[1];
    const float* poses   = (const float*)d_in[2];
    const float* intr    = (const float*)d_in[3];
    /* d_in[4] grid_points_3d: unused by reference */
    const float* patch_w = (const float*)d_in[5];
    const float* patch_b = (const float*)d_in[6];
    const float* cls_tok = (const float*)d_in[7];
    const float* pos_emb = (const float*)d_in[8];
    const float* ln1_g   = (const float*)d_in[9];
    const float* ln1_b   = (const float*)d_in[10];
    const float* qkv_w   = (const float*)d_in[11];
    const float* qkv_b   = (const float*)d_in[12];
    const float* proj_w  = (const float*)d_in[13];
    const float* proj_b  = (const float*)d_in[14];
    const float* ln2_g   = (const float*)d_in[15];
    const float* ln2_b   = (const float*)d_in[16];
    const float* fc1_w   = (const float*)d_in[17];
    const float* fc1_b   = (const float*)d_in[18];
    const float* fc2_w   = (const float*)d_in[19];
    const float* fc2_b   = (const float*)d_in[20];
    const float* nf_g    = (const float*)d_in[21];
    const float* nf_b    = (const float*)d_in[22];
    const float* conv_w  = (const float*)d_in[23];
    const float* conv_b  = (const float*)d_in[24];

    // all activation buffers padded to MPAD (=1664) rows -> branch-free GEMM.
    // pad rows hold garbage; they only ever produce pad rows, never real ones.
    float* ws = (float*)d_ws;
    size_t off = 0;
    float* TOK = ws + off; off += (size_t)MPAD * 768;     // tokens (8,197,768)+pad
    float* H   = ws + off; off += (size_t)MPAD * 768;     // LN output
    float* QKV = ws + off; off += (size_t)MPAD * 2304;    // qkv / (alias) im2col A
    float* ATT = ws + off; off += (size_t)MPAD * 768;     // attention out
    float* MLP = ws + off; off += (size_t)MPAD * 3072;    // fc1 out / (alias) Ptok
    float* C2W  = ws + off; off += 128;
    float* D14  = ws + off; off += 1568;
    float* SSUM = ws + off; off += (size_t)392 * 768;     // SSUM/SMAX/CNT contiguous
    float* SMAX = ws + off; off += (size_t)392 * 768;
    float* CNT  = ws + off; off += 392;

    float* APATCH = QKV;   // 1664*768 fits in QKV region
    float* PTOK   = MLP;   // 1664*768 fits in MLP region

    // ---- patch embedding
    im2col_kernel<<<1568, 256, 0, stream>>>(imgs, APATCH);
    gemm_wmma_kernel<<<dim3(6, 13), 256, 0, stream>>>(
        APATCH, patch_w, patch_b, nullptr, PTOK, 768, 768, 0);
    assemble_tok_kernel<<<1576, 256, 0, stream>>>(PTOK, cls_tok, pos_emb, TOK);

    // ---- 12 transformer blocks
    for (int l = 0; l < 12; ++l) {
        ln_kernel<<<1576, 256, 0, stream>>>(TOK, ln1_g + l * 768, ln1_b + l * 768, H);
        gemm_wmma_kernel<<<dim3(18, 13), 256, 0, stream>>>(
            H, qkv_w + (size_t)l * 768 * 2304, qkv_b + (size_t)l * 2304,
            nullptr, QKV, 2304, 768, 0);
        attn_kernel<<<dim3(197, 12, 8), 64, 0, stream>>>(QKV, ATT);
        gemm_wmma_kernel<<<dim3(6, 13), 256, 0, stream>>>(
            ATT, proj_w + (size_t)l * 768 * 768, proj_b + (size_t)l * 768,
            TOK, TOK, 768, 768, 0);
        ln_kernel<<<1576, 256, 0, stream>>>(TOK, ln2_g + l * 768, ln2_b + l * 768, H);
        gemm_wmma_kernel<<<dim3(24, 13), 256, 0, stream>>>(
            H, fc1_w + (size_t)l * 768 * 3072, fc1_b + (size_t)l * 3072,
            nullptr, MLP, 3072, 768, 1);
        gemm_wmma_kernel<<<dim3(6, 13), 256, 0, stream>>>(
            MLP, fc2_w + (size_t)l * 3072 * 768, fc2_b + (size_t)l * 768,
            TOK, TOK, 768, 3072, 0);
    }

    // ---- final LN
    ln_kernel<<<1576, 256, 0, stream>>>(TOK, nf_g, nf_b, H);

    // ---- geometry + BEV scatter
    inv4_kernel<<<1, 8, 0, stream>>>(poses, C2W);
    depth_resize_kernel<<<8, 256, 0, stream>>>(depths, D14);
    int nz = 392 * 768 * 2 + 392;
    zero_fill_kernel<<<(nz + 255) / 256, 256, 0, stream>>>(SSUM, nz);
    scatter_kernel<<<1568, 256, 0, stream>>>(H, D14, intr, C2W, SSUM, SMAX, CNT);

    // ---- merged bev + 1x1 conv
    bev_conv_kernel<<<392, 256, 0, stream>>>(SSUM, SMAX, CNT, conv_w, conv_b,
                                             (float*)d_out);
}
